// TripletCenterCosineLoss_15917148799621
// MI455X (gfx1250) — matrix operand order repaired and checked by reading, not compiled, not run on hardware
//
#include <hip/hip_runtime.h>
#include <hip/hip_bf16.h>

typedef __attribute__((ext_vector_type(16))) __bf16       v16bf;
typedef __attribute__((ext_vector_type(8)))  float        v8f;
typedef __attribute__((ext_vector_type(4)))  unsigned int u32x4;

#define B_N          65536
#define C_N          1000
#define C_PAD        1024
#define D_K          512
#define ROWS_PER_BLK 64
#define LDS_PITCH    520   // D_K + 8 ushorts -> 4-bank stagger per row, conflict-free b128 frag loads

static __device__ __forceinline__ unsigned short f32_bf16_rne(float f) {
  unsigned int u = __float_as_uint(f);
  u += 0x7FFFu + ((u >> 16) & 1u);          // round-to-nearest-even
  return (unsigned short)(u >> 16);
}
static __device__ __forceinline__ float bf16_f32(unsigned short h) {
  return __uint_as_float(((unsigned int)h) << 16);
}

// ---------------- Kernel 1: normalize centers, split to bf16 hi/lo, pad to 1024 rows ----------
__global__ void __launch_bounds__(128)
prep_centers(const float* __restrict__ centers,
             unsigned short* __restrict__ ch,
             unsigned short* __restrict__ cl) {
  __shared__ float red[128];
  const int r = blockIdx.x;   // 0..1023
  const int t = threadIdx.x;  // 0..127
  if (r >= C_N) {             // zero padding rows (masked in epilogue anyway)
    for (int i = t; i < D_K; i += 128) { ch[(size_t)r * D_K + i] = 0; cl[(size_t)r * D_K + i] = 0; }
    return;
  }
  const float* row = centers + (size_t)r * D_K;
  float vals[4];
  float ss = 0.f;
#pragma unroll
  for (int j = 0; j < 4; ++j) { float v = row[t + j * 128]; vals[j] = v; ss += v * v; }
  red[t] = ss;
  __syncthreads();
  for (int s = 64; s > 0; s >>= 1) { if (t < s) red[t] += red[t + s]; __syncthreads(); }
  const float rn = 1.0f / sqrtf(red[0]);
#pragma unroll
  for (int j = 0; j < 4; ++j) {
    float v = vals[j] * rn;
    unsigned short h = f32_bf16_rne(v);
    unsigned short l = f32_bf16_rne(v - bf16_f32(h));
    ch[(size_t)r * D_K + t + j * 128] = h;
    cl[(size_t)r * D_K + t + j * 128] = l;
  }
}

// ---------------- Kernel 2: fused GEMM (bf16x3 WMMA) + per-row min/label-gather + hinge -------
__global__ void __launch_bounds__(256)
triplet_main(const float* __restrict__ x,
             const int*   __restrict__ labels,
             const unsigned short* __restrict__ ch,
             const unsigned short* __restrict__ cl,
             float* __restrict__ partials) {
  extern __shared__ char smem[];
  unsigned short* xh     = (unsigned short*)smem;                      // [64][520]
  unsigned short* xl     = xh + ROWS_PER_BLK * LDS_PITCH;              // [64][520]
  int*            slab   = (int*)(xl + ROWS_PER_BLK * LDS_PITCH);      // [64]
  float*          rmin_s = (float*)(slab + ROWS_PER_BLK);              // [2][64]
  float*          rpos_s = rmin_s + 2 * ROWS_PER_BLK;                  // [2][64]
  float*          blkred = rpos_s + 2 * ROWS_PER_BLK;                  // [8]

  const int tid  = threadIdx.x;
  const int lane = tid & 31;
  const int wid  = tid >> 5;
  const int row0 = blockIdx.x * ROWS_PER_BLK;

  // ---- stage 64x512 f32 x-tile into LDS as bf16 hi/lo (coalesced b128 reads) ----
  const float4* xv = (const float4*)(x + (size_t)row0 * D_K);
#pragma unroll 4
  for (int it = 0; it < (ROWS_PER_BLK * D_K / 4) / 256; ++it) {
    const int idx4 = it * 256 + tid;       // float4 index within the contiguous tile
    const int r = idx4 >> 7;               // 128 float4 per row
    const int c = (idx4 & 127) << 2;
    const float4 v = xv[idx4];
    const float f[4] = {v.x, v.y, v.z, v.w};
#pragma unroll
    for (int j = 0; j < 4; ++j) {
      unsigned short h = f32_bf16_rne(f[j]);
      xh[r * LDS_PITCH + c + j] = h;
      xl[r * LDS_PITCH + c + j] = f32_bf16_rne(f[j] - bf16_f32(h));
    }
  }
  if (tid < ROWS_PER_BLK) slab[tid] = labels[row0 + tid];
  __syncthreads();

  // wave -> (M-tile, column half): 4 M-tiles x 2 halves
  const int mt    = wid & 3;
  const int half  = wid >> 2;
  const int rowb  = mt * 16;
  const int nlane = lane & 15;
  const int sel   = lane >> 4;

  int lab[8];
#pragma unroll
  for (int v = 0; v < 8; ++v) lab[v] = slab[rowb + v + sel * 8];  // C/D reg v: rows v / v+8

  float rmin[8], rpos[8];
#pragma unroll
  for (int v = 0; v < 8; ++v) { rmin[v] = 1e30f; rpos[v] = 0.f; }

  // A-fragment addressing per ISA 16-bit A layout (lane&15 = row; sel picks K sub-block)
  const unsigned short* xh_row = xh + (rowb + nlane) * LDS_PITCH;
  const unsigned short* xl_row = xl + (rowb + nlane) * LDS_PITCH;
  const int aoff1 = sel * 8;
  const int aoff2 = 16 + sel * 8;

  // two N-tiles per iteration: A-fragments amortized 2x, two independent accumulators for ILP
  for (int nt = 0; nt < 32; nt += 2) {
    const int cg0 = half * 512 + nt * 16 + nlane;   // center column owned by this lane, tile 0
    const int cg1 = cg0 + 16;                       // tile 1
    // B-fragment: lane&15 = column; lanes 0-15 hold K=kb..kb+7,kb+16..kb+23, lanes 16-31 K=kb+8..kb+15,kb+24..kb+31
    const unsigned short* bh0 = ch + (size_t)cg0 * D_K + sel * 8;
    const unsigned short* bl0 = cl + (size_t)cg0 * D_K + sel * 8;
    const unsigned short* bh1 = ch + (size_t)cg1 * D_K + sel * 8;
    const unsigned short* bl1 = cl + (size_t)cg1 * D_K + sel * 8;

    v8f acc0 = {0.f, 0.f, 0.f, 0.f, 0.f, 0.f, 0.f, 0.f};
    v8f acc1 = {0.f, 0.f, 0.f, 0.f, 0.f, 0.f, 0.f, 0.f};
#pragma unroll 2
    for (int kb = 0; kb < D_K; kb += 32) {
      union Frag { v16bf v; u32x4 q[2]; } ah, al, b0h, b0l, b1h, b1l;
      ah.q[0]  = *(const u32x4*)(xh_row + kb + aoff1);
      ah.q[1]  = *(const u32x4*)(xh_row + kb + aoff2);
      al.q[0]  = *(const u32x4*)(xl_row + kb + aoff1);
      al.q[1]  = *(const u32x4*)(xl_row + kb + aoff2);
      b0h.q[0] = *(const u32x4*)(bh0 + kb);
      b0h.q[1] = *(const u32x4*)(bh0 + kb + 16);
      b0l.q[0] = *(const u32x4*)(bl0 + kb);
      b0l.q[1] = *(const u32x4*)(bl0 + kb + 16);
      b1h.q[0] = *(const u32x4*)(bh1 + kb);
      b1h.q[1] = *(const u32x4*)(bh1 + kb + 16);
      b1l.q[0] = *(const u32x4*)(bl1 + kb);
      b1l.q[1] = *(const u32x4*)(bl1 + kb + 16);
      // bf16x3 split precision; interleave the two tiles so consecutive WMMAs are independent
      acc0 = __builtin_amdgcn_wmma_f32_16x16x32_bf16(false, ah.v, false, b0h.v, (short)0, acc0, false, false);
      acc1 = __builtin_amdgcn_wmma_f32_16x16x32_bf16(false, ah.v, false, b1h.v, (short)0, acc1, false, false);
      acc0 = __builtin_amdgcn_wmma_f32_16x16x32_bf16(false, ah.v, false, b0l.v, (short)0, acc0, false, false);
      acc1 = __builtin_amdgcn_wmma_f32_16x16x32_bf16(false, ah.v, false, b1l.v, (short)0, acc1, false, false);
      acc0 = __builtin_amdgcn_wmma_f32_16x16x32_bf16(false, al.v, false, b0h.v, (short)0, acc0, false, false);
      acc1 = __builtin_amdgcn_wmma_f32_16x16x32_bf16(false, al.v, false, b1h.v, (short)0, acc1, false, false);
    }
    // fused epilogue, fully branchless: cndmask/min/add only, no exec-mask manipulation
#pragma unroll
    for (int v = 0; v < 8; ++v) {
      const float d0   = 1.0f - acc0[v];
      const float d1   = 1.0f - acc1[v];
      const bool  hit0 = (cg0 == lab[v]);             // label column: goes to pos, not min
      const bool  hit1 = (cg1 == lab[v]);
      rpos[v] += hit0 ? d0 : 0.0f;
      rpos[v] += hit1 ? d1 : 0.0f;
      const float m0 = (hit0 || (cg0 >= C_N)) ? 1e30f : d0;   // mask label + padding columns
      const float m1 = (hit1 || (cg1 >= C_N)) ? 1e30f : d1;
      rmin[v] = fminf(rmin[v], fminf(m0, m1));
    }
  }

  // reduce across the 16 lanes of each half (masks 1,2,4,8 stay within lane groups 0-15 / 16-31)
#pragma unroll
  for (int m = 1; m <= 8; m <<= 1) {
#pragma unroll
    for (int v = 0; v < 8; ++v) {
      rmin[v] = fminf(rmin[v], __shfl_xor(rmin[v], m, 32));
      rpos[v] += __shfl_xor(rpos[v], m, 32);
    }
  }
  if (nlane == 0) {                       // lanes 0 (rows rowb..rowb+7) and 16 (rows rowb+8..+15)
    const int rb = rowb + sel * 8;
#pragma unroll
    for (int v = 0; v < 8; ++v) {
      rmin_s[half * ROWS_PER_BLK + rb + v] = rmin[v];
      rpos_s[half * ROWS_PER_BLK + rb + v] = rpos[v];
    }
  }
  __syncthreads();

  // combine column halves, hinge, block-sum (deterministic, no float atomics)
  float h = 0.f;
  if (tid < ROWS_PER_BLK) {
    const float neg = fminf(rmin_s[tid], rmin_s[ROWS_PER_BLK + tid]);
    const float pos = rpos_s[tid] + rpos_s[ROWS_PER_BLK + tid];
    h = fmaxf(0.f, pos + 1.0f - neg);
  }
#pragma unroll
  for (int m = 1; m <= 16; m <<= 1) h += __shfl_xor(h, m, 32);
  if (lane == 0) blkred[wid] = h;
  __syncthreads();
  if (tid == 0) {
    float s = 0.f;
#pragma unroll
    for (int w = 0; w < 8; ++w) s += blkred[w];
    partials[blockIdx.x] = s;
  }
}

// ---------------- Kernel 3: deterministic fixed-order reduction of 1024 partials --------------
__global__ void __launch_bounds__(256)
finalize(const float* __restrict__ partials, float* __restrict__ out) {
  __shared__ float red[256];
  const int t = threadIdx.x;
  red[t] = partials[t] + partials[t + 256] + partials[t + 512] + partials[t + 768];
  __syncthreads();
  for (int s = 128; s > 0; s >>= 1) { if (t < s) red[t] += red[t + s]; __syncthreads(); }
  if (t == 0) out[0] = red[0] * (1.0f / (float)B_N);
}

extern "C" void kernel_launch(void* const* d_in, const int* in_sizes, int n_in,
                              void* d_out, int out_size, void* d_ws, size_t ws_size,
                              hipStream_t stream) {
  (void)in_sizes; (void)n_in; (void)out_size; (void)ws_size;
  const float* x       = (const float*)d_in[0];
  const int*   labels  = (const int*)d_in[1];
  const float* centers = (const float*)d_in[2];

  unsigned short* ch = (unsigned short*)d_ws;                 // 1 MB
  unsigned short* cl = ch + (size_t)C_PAD * D_K;              // 1 MB
  float* partials    = (float*)(cl + (size_t)C_PAD * D_K);    // 4 KB

  prep_centers<<<C_PAD, 128, 0, stream>>>(centers, ch, cl);

  const size_t smem = (size_t)2 * ROWS_PER_BLK * LDS_PITCH * sizeof(unsigned short)
                    + ROWS_PER_BLK * sizeof(int)
                    + 4 * ROWS_PER_BLK * sizeof(float)
                    + 8 * sizeof(float);                      // ~134.4 KB (< 320 KB WGP limit)
  triplet_main<<<B_N / ROWS_PER_BLK, 256, smem, stream>>>(x, labels, ch, cl, partials);

  finalize<<<1, 256, 0, stream>>>(partials, (float*)d_out);
}
